// EGNNScoreNet_3212635537410
// MI455X (gfx1250) — matrix operand order repaired and checked by reading, not compiled, not run on hardware
//
#include <hip/hip_runtime.h>
#include <stdint.h>

// ---------------- problem constants ----------------
#define BATCH 4
#define NPTS  5000
#define SPD   3
#define KNN   20
#define NEDGE (NPTS*KNN)      // 100000
#define DH    64
#define DCOND 2
#define DC    34
#define H4    (4*DC)          // 136
#define TSTEPS 4
#define KPAD_E 160            // 2*DH+1 = 129 -> pad to 5*32
#define KPAD_H 192            // 2*DH+DC = 162 -> pad to 6*32
#define WAVES 4
#define NTILE_E (NEDGE/16)    // 6250
#define NTILE_N ((NPTS+15)/16) // 313
#define KNN_T 128

// packed weight layout per step (bf16 element offsets)
#define OFF_WE1 0
#define OFF_WE2 10240
#define OFF_WX1 14336
#define OFF_WH1 18432
#define OFF_WH2 30720
#define STEP_PACK 34816       // bf16 elements per step

typedef __attribute__((ext_vector_type(16))) __bf16 v16bf;
typedef __attribute__((ext_vector_type(8)))  float  v8f;

// ---------------- device helpers ----------------
__device__ __forceinline__ uint32_t f2bf(float x) {
  union { float f; uint32_t u; } c; c.f = x;
  uint32_t r = c.u + 0x7FFFu + ((c.u >> 16) & 1u);
  return r >> 16;
}
__device__ __forceinline__ float bf2f(uint16_t b) {
  union { uint32_t u; float f; } c; c.u = ((uint32_t)b) << 16;
  return c.f;
}
__device__ __forceinline__ float silu_f(float x) { return x / (1.f + expf(-x)); }
__device__ __forceinline__ float gelu_f(float x) {
  float x3 = x * x * x;
  return 0.5f * x * (1.f + tanhf(0.7978845608028654f * (x + 0.044715f * x3)));
}

// A fragment (16x32 bf16) from LDS tile stored row-major [16][ldk] as bf16(u16).
// ISA layout: M = lane%16 ; K = ks*32 + (lane>=16 ? 8 : 0) + (j<8 ? j : j+8)
__device__ __forceinline__ v16bf load_a_frag(const uint16_t* tile, int ldk, int ks, int lane) {
  int m  = lane & 15;
  int kb = ks * 32 + ((lane & 16) ? 8 : 0);
  const uint16_t* row = tile + m * ldk;
  union { v16bf v; uint32_t u[8]; } r;
#pragma unroll
  for (int p = 0; p < 8; ++p) {
    int j0 = 2 * p;
    int k0 = kb + (j0 < 8 ? j0 : j0 + 8);
    r.u[p] = (uint32_t)row[k0] | ((uint32_t)row[k0 + 1] << 16);
  }
  return r.v;
}

// B fragment: pre-packed so each lane reads 8 contiguous dwords.
__device__ __forceinline__ v16bf load_b_frag(const uint32_t* fragBaseDw, int lane) {
  union { v16bf v; uint32_t u[8]; } r;
  const uint32_t* p = fragBaseDw + lane * 8;
#pragma unroll
  for (int i = 0; i < 8; ++i) r.u[i] = p[i];
  return r.v;
}

__device__ __forceinline__ v8f wmma_bf16(v16bf a, v16bf b, v8f c) {
  return __builtin_amdgcn_wmma_f32_16x16x32_bf16(false, a, false, b, (short)0, c, false, false);
}

// ---------------- weight packing ----------------
// B-fragment layout: lane L holds col N = ns*16 + L%16, rows K = ks*32 + (L>=16?8:0) + (j<8 ? j : j+8)
__global__ void pack_weights_kernel(const float* __restrict__ We1, const float* __restrict__ We2,
                                    const float* __restrict__ Wx1, const float* __restrict__ Wh1,
                                    const float* __restrict__ Wh2, uint32_t* __restrict__ pw) {
  int gid = blockIdx.x * blockDim.x + threadIdx.x;
  const int total = TSTEPS * 68 * 32;
  if (gid >= total) return;
  int lane = gid & 31;
  int f    = (gid >> 5) % 68;
  int s    = gid / (68 * 32);
  const float* src; int Ksrc, ks, ns, dstOff;
  if (f < 20)      { int fl = f;      src = We1 + s*129*64; Ksrc = 129; ks = fl/4; ns = fl%4; dstOff = OFF_WE1 + fl*512; }
  else if (f < 28) { int fl = f - 20; src = We2 + s*64*64;  Ksrc = 64;  ks = fl/4; ns = fl%4; dstOff = OFF_WE2 + fl*512; }
  else if (f < 36) { int fl = f - 28; src = Wx1 + s*64*64;  Ksrc = 64;  ks = fl/4; ns = fl%4; dstOff = OFF_WX1 + fl*512; }
  else if (f < 60) { int fl = f - 36; src = Wh1 + s*162*64; Ksrc = 162; ks = fl/4; ns = fl%4; dstOff = OFF_WH1 + fl*512; }
  else             { int fl = f - 60; src = Wh2 + s*64*64;  Ksrc = 64;  ks = fl/4; ns = fl%4; dstOff = OFF_WH2 + fl*512; }
  int n = ns*16 + (lane & 15);
  int kHalf = (lane & 16) ? 8 : 0;
  uint32_t outv[8];
#pragma unroll
  for (int p = 0; p < 8; ++p) {
    int k0 = ks*32 + kHalf + (2*p < 8 ? 2*p : 2*p + 8);
    float a = (k0     < Ksrc) ? src[(size_t)k0*64 + n]       : 0.f;
    float b = (k0 + 1 < Ksrc) ? src[(size_t)(k0+1)*64 + n]   : 0.f;
    outv[p] = f2bf(a) | (f2bf(b) << 16);
  }
  uint32_t* dst = pw + ((size_t)s*STEP_PACK + dstOff + lane*16) / 2;
#pragma unroll
  for (int p = 0; p < 8; ++p) dst[p] = outv[p];
}

// ---------------- conditioning MLP ----------------
__global__ void cond_kernel(const float* __restrict__ t, const float* __restrict__ conditioning,
                            const float* __restrict__ Wc1, const float* __restrict__ bc1,
                            const float* __restrict__ Wc2, const float* __restrict__ bc2,
                            const float* __restrict__ Wc3, const float* __restrict__ bc3,
                            float* __restrict__ condW) {
  __shared__ float c0[BATCH*DC];
  __shared__ float c1[BATCH*H4];
  __shared__ float c2[BATCH*H4];
  int tid = threadIdx.x;
  const float LOG1E4 = 9.210340371976184f;
  for (int idx = tid; idx < BATCH*DC; idx += blockDim.x) {
    int b = idx / DC, i = idx % DC;
    float v;
    if (i < 16)      v = sinf(t[b] * expf(-LOG1E4 * (float)i / 16.f));
    else if (i < 32) v = cosf(t[b] * expf(-LOG1E4 * (float)(i - 16) / 16.f));
    else             v = conditioning[b*DCOND + (i - 32)];
    c0[idx] = v;
  }
  __syncthreads();
  for (int idx = tid; idx < BATCH*H4; idx += blockDim.x) {
    int b = idx / H4, u = idx % H4;
    float s = bc1[u];
    for (int k = 0; k < DC; ++k) s += c0[b*DC + k] * Wc1[k*H4 + u];
    c1[idx] = gelu_f(s);
  }
  __syncthreads();
  for (int idx = tid; idx < BATCH*H4; idx += blockDim.x) {
    int b = idx / H4, u = idx % H4;
    float s = bc2[u];
    for (int k = 0; k < H4; ++k) s += c1[b*H4 + k] * Wc2[k*H4 + u];
    c2[idx] = gelu_f(s);
  }
  __syncthreads();
  for (int idx = tid; idx < BATCH*DC; idx += blockDim.x) {
    int b = idx / DC, u = idx % DC;
    float s = bc3[u];
    for (int k = 0; k < H4; ++k) s += c2[b*H4 + k] * Wc3[k*DC + u];
    condW[idx] = s;
  }
}

// ---------------- kNN ----------------
__global__ void knn_kernel(const float* __restrict__ z, const float* __restrict__ mask,
                           int* __restrict__ tgtIdx) {
  __shared__ float cxs[KNN_T*SPD];
  __shared__ float cms[KNN_T];
  __shared__ float dl[KNN_T*KNN];
  __shared__ int   il[KNN_T*KNN];
  int tid = threadIdx.x;
  int q = blockIdx.x * KNN_T + tid;
  int b = blockIdx.y;
  bool qv = q < NPTS;
  float qx = 0.f, qy = 0.f, qz = 0.f, qm = 0.f;
  if (qv) {
    const float* p = z + ((size_t)b*NPTS + q)*SPD;
    qx = p[0]; qy = p[1]; qz = p[2];
    qm = mask[b*NPTS + q];
  }
  int base = tid * KNN;
  for (int k = 0; k < KNN; ++k) { dl[base + k] = 3.0e38f; il[base + k] = 0; }
  for (int c0 = 0; c0 < NPTS; c0 += KNN_T) {
    __syncthreads();
    int j = c0 + tid;
    if (j < NPTS) {
      const float* p = z + ((size_t)b*NPTS + j)*SPD;
      cxs[tid*SPD+0] = p[0]; cxs[tid*SPD+1] = p[1]; cxs[tid*SPD+2] = p[2];
      cms[tid] = mask[b*NPTS + j];
    }
    __syncthreads();
    if (qv) {
      int lim = NPTS - c0; if (lim > KNN_T) lim = KNN_T;
      for (int jj = 0; jj < lim; ++jj) {
        if (qm * cms[jj] <= 0.f) continue;
        float dx = qx - cxs[jj*SPD+0];
        float dy = qy - cxs[jj*SPD+1];
        float dz = qz - cxs[jj*SPD+2];
        float d2 = dx*dx + dy*dy + dz*dz;
        if (d2 < dl[base + KNN - 1]) {
          int p = KNN - 1;
          while (p > 0 && dl[base + p - 1] > d2) {
            dl[base + p] = dl[base + p - 1];
            il[base + p] = il[base + p - 1];
            --p;
          }
          dl[base + p] = d2;
          il[base + p] = c0 + jj;
        }
      }
    }
  }
  if (qv) {
    for (int k = 0; k < KNN; ++k)
      tgtIdx[(size_t)b*NEDGE + (size_t)q*KNN + k] = il[base + k];
  }
}

// ---------------- init / zero / out ----------------
__global__ void init_kernel(const float* __restrict__ z, const float* __restrict__ Wemb,
                            const float* __restrict__ bemb,
                            float* __restrict__ h, float* __restrict__ x) {
  int gid = blockIdx.x * blockDim.x + threadIdx.x;
  const int hTot = BATCH*NPTS*DH;
  const int xTot = BATCH*NPTS*SPD;
  if (gid < hTot) {
    int c = gid % DH;
    h[gid] = Wemb[c] + bemb[c];
  } else if (gid < hTot + xTot) {
    int i = gid - hTot;
    x[i] = z[i];
  }
}

__global__ void zero_kernel(float* __restrict__ agg, float* __restrict__ dxb) {
  int gid = blockIdx.x * blockDim.x + threadIdx.x;
  const int aTot = BATCH*NPTS*DH;
  const int dTot = BATCH*NPTS*SPD;
  if (gid < aTot) agg[gid] = 0.f;
  else if (gid < aTot + dTot) dxb[gid - aTot] = 0.f;
}

__global__ void out_kernel(const float* __restrict__ z, const float* __restrict__ mask,
                           const float* __restrict__ x, float* __restrict__ out) {
  int gid = blockIdx.x * blockDim.x + threadIdx.x;
  if (gid < BATCH*NPTS*SPD) {
    int n = (gid / SPD) % NPTS;
    int b = gid / (NPTS*SPD);
    out[gid] = (x[gid] - z[gid]) * mask[b*NPTS + n];
  }
}

// ---------------- edge kernel (WMMA heavy) ----------------
struct EdgeLds {
  uint16_t a[16*KPAD_E];   // e_in tile, bf16
  uint16_t m1[16*DH];      // silu(layer1)
  uint16_t m2[16*DH];      // msg = silu(layer2)*edge_mask
  uint16_t tt[16*DH];      // silu(msg@Wx1+bx1)
  float    xd[16*SPD];
  float    emask[16];
  float    xw[16];
  int      tgt[16];
};

__global__ void edge_kernel(const float* __restrict__ x, const float* __restrict__ h,
                            const float* __restrict__ mask, const int* __restrict__ tgtIdx,
                            const uint32_t* __restrict__ pw,
                            const float* __restrict__ be1, const float* __restrict__ be2,
                            const float* __restrict__ bx1, const float* __restrict__ wx2,
                            const float* __restrict__ bx2,
                            float* __restrict__ agg, float* __restrict__ dxb, int step) {
  __shared__ EdgeLds lds[WAVES];
  int lane = threadIdx.x & 31;
  int wid  = threadIdx.x >> 5;
  int tile = blockIdx.x * WAVES + wid;
  int b    = blockIdx.y;
  bool active = tile < NTILE_E;
  EdgeLds& L = lds[wid];
  const uint32_t* pws = pw + (size_t)step * (STEP_PACK/2);

  // phase A: build 16x160 bf16 feature tile [h_src | h_tgt | d2 | 0-pad]
  if (active) {
    int m = lane >> 1;
    int e = tile*16 + m;
    int sn = e / KNN;
    int tn = tgtIdx[(size_t)b*NEDGE + e];
    if ((lane & 1) == 0) {
      L.tgt[m] = tn;
      L.emask[m] = mask[b*NPTS + sn] * mask[b*NPTS + tn];
      float d2 = 0.f;
#pragma unroll
      for (int d = 0; d < SPD; ++d) {
        float xd = x[((size_t)b*NPTS + sn)*SPD + d] - x[((size_t)b*NPTS + tn)*SPD + d];
        L.xd[m*SPD + d] = xd;
        d2 += xd * xd;
      }
      L.a[m*KPAD_E + 128] = (uint16_t)f2bf(d2);
      for (int k = 129; k < KPAD_E; ++k) L.a[m*KPAD_E + k] = 0;
      const float* hs = h + ((size_t)b*NPTS + sn)*DH;
      for (int c = 0; c < DH; ++c) L.a[m*KPAD_E + c] = (uint16_t)f2bf(hs[c]);
    } else {
      const float* ht = h + ((size_t)b*NPTS + tn)*DH;
      for (int c = 0; c < DH; ++c) L.a[m*KPAD_E + DH + c] = (uint16_t)f2bf(ht[c]);
    }
  }
  __syncthreads();

  int row0 = (lane & 16) ? 8 : 0;
  int colc = lane & 15;

  // layer 1: [16x160] @ We1[160x64] -> silu
  if (active) {
    v16bf af[5];
#pragma unroll
    for (int ks = 0; ks < 5; ++ks) af[ks] = load_a_frag(L.a, KPAD_E, ks, lane);
#pragma unroll
    for (int ns = 0; ns < 4; ++ns) {
      float bias = be1[step*DH + ns*16 + colc];
      v8f c;
#pragma unroll
      for (int r = 0; r < 8; ++r) c[r] = bias;
#pragma unroll
      for (int ks = 0; ks < 5; ++ks)
        c = wmma_bf16(af[ks], load_b_frag(pws + (OFF_WE1 + (ks*4+ns)*512)/2, lane), c);
#pragma unroll
      for (int r = 0; r < 8; ++r)
        L.m1[(row0+r)*DH + ns*16 + colc] = (uint16_t)f2bf(silu_f(c[r]));
    }
  }
  __syncthreads();

  // layer 2: @ We2 -> silu -> * edge_mask
  if (active) {
    v16bf af[2];
#pragma unroll
    for (int ks = 0; ks < 2; ++ks) af[ks] = load_a_frag(L.m1, DH, ks, lane);
#pragma unroll
    for (int ns = 0; ns < 4; ++ns) {
      float bias = be2[step*DH + ns*16 + colc];
      v8f c;
#pragma unroll
      for (int r = 0; r < 8; ++r) c[r] = bias;
#pragma unroll
      for (int ks = 0; ks < 2; ++ks)
        c = wmma_bf16(af[ks], load_b_frag(pws + (OFF_WE2 + (ks*4+ns)*512)/2, lane), c);
#pragma unroll
      for (int r = 0; r < 8; ++r) {
        int row = row0 + r;
        L.m2[row*DH + ns*16 + colc] = (uint16_t)f2bf(silu_f(c[r]) * L.emask[row]);
      }
    }
  }
  __syncthreads();

  // layer 3: msg @ Wx1 -> silu
  if (active) {
    v16bf af[2];
#pragma unroll
    for (int ks = 0; ks < 2; ++ks) af[ks] = load_a_frag(L.m2, DH, ks, lane);
#pragma unroll
    for (int ns = 0; ns < 4; ++ns) {
      float bias = bx1[step*DH + ns*16 + colc];
      v8f c;
#pragma unroll
      for (int r = 0; r < 8; ++r) c[r] = bias;
#pragma unroll
      for (int ks = 0; ks < 2; ++ks)
        c = wmma_bf16(af[ks], load_b_frag(pws + (OFF_WX1 + (ks*4+ns)*512)/2, lane), c);
#pragma unroll
      for (int r = 0; r < 8; ++r)
        L.tt[(row0+r)*DH + ns*16 + colc] = (uint16_t)f2bf(silu_f(c[r]));
    }
  }
  __syncthreads();

  // xw = tt . Wx2 + bx2
  if (active && lane < 16) {
    float s = bx2[step];
    const float* w = wx2 + step*DH;
    for (int c = 0; c < DH; ++c) s += bf2f(L.tt[lane*DH + c]) * w[c];
    L.xw[lane] = s;
  }
  __syncthreads();

  // scatter: agg[tgt] += msg ; dx[tgt] += xd * xw * edge_mask (L2-resident f32 atomics)
  if (active) {
#pragma unroll 1
    for (int m = 0; m < 16; ++m) {
      int tg = L.tgt[m];
      float* ag = agg + ((size_t)b*NPTS + tg)*DH;
      atomicAdd(&ag[lane],      bf2f(L.m2[m*DH + lane]));
      atomicAdd(&ag[lane + 32], bf2f(L.m2[m*DH + lane + 32]));
    }
    if (lane < 16) {
      int tg = L.tgt[lane];
      float w = L.xw[lane] * L.emask[lane];
      float* dxp = dxb + ((size_t)b*NPTS + tg)*SPD;
#pragma unroll
      for (int d = 0; d < SPD; ++d) atomicAdd(&dxp[d], L.xd[lane*SPD + d] * w);
    }
  }
}

// ---------------- node kernel (WMMA) ----------------
struct NodeLds {
  uint16_t a[16*KPAD_H];   // [h | agg | cond | 0-pad] bf16
  uint16_t m1[16*DH];
  float    nmask[16];
};

__global__ void node_kernel(float* __restrict__ x, float* __restrict__ h,
                            const float* __restrict__ mask,
                            const float* __restrict__ agg, const float* __restrict__ dxb,
                            const float* __restrict__ condW,
                            const uint32_t* __restrict__ pw,
                            const float* __restrict__ bh1, const float* __restrict__ bh2,
                            int step) {
  __shared__ NodeLds lds[WAVES];
  int lane = threadIdx.x & 31;
  int wid  = threadIdx.x >> 5;
  int tile = blockIdx.x * WAVES + wid;
  int b    = blockIdx.y;
  bool active = tile < NTILE_N;
  NodeLds& L = lds[wid];
  const uint32_t* pws = pw + (size_t)step * (STEP_PACK/2);
  int n0 = tile * 16;

  if (active) {
    int m = lane >> 1;
    int n = n0 + m;
    bool valid = n < NPTS;
    const float* hp = h + ((size_t)b*NPTS + n)*DH;
    const float* ap = agg + ((size_t)b*NPTS + n)*DH;
    if ((lane & 1) == 0) {
      L.nmask[m] = valid ? mask[b*NPTS + n] : 0.f;
      for (int c = 0; c < 96; ++c) {
        float v = 0.f;
        if (valid) v = (c < DH) ? hp[c] : ap[c - DH];
        L.a[m*KPAD_H + c] = (uint16_t)f2bf(v);
      }
    } else {
      for (int c = 96; c < KPAD_H; ++c) {
        float v = 0.f;
        if (c < 128)            { if (valid) v = ap[c - DH]; }
        else if (c < 128 + DC)  v = condW[b*DC + (c - 128)];
        L.a[m*KPAD_H + c] = (uint16_t)f2bf(v);
      }
    }
  }
  __syncthreads();

  int row0 = (lane & 16) ? 8 : 0;
  int colc = lane & 15;

  // layer 1: [16x192] @ Wh1 -> silu
  if (active) {
    v16bf af[6];
#pragma unroll
    for (int ks = 0; ks < 6; ++ks) af[ks] = load_a_frag(L.a, KPAD_H, ks, lane);
#pragma unroll
    for (int ns = 0; ns < 4; ++ns) {
      float bias = bh1[step*DH + ns*16 + colc];
      v8f c;
#pragma unroll
      for (int r = 0; r < 8; ++r) c[r] = bias;
#pragma unroll
      for (int ks = 0; ks < 6; ++ks)
        c = wmma_bf16(af[ks], load_b_frag(pws + (OFF_WH1 + (ks*4+ns)*512)/2, lane), c);
#pragma unroll
      for (int r = 0; r < 8; ++r)
        L.m1[(row0+r)*DH + ns*16 + colc] = (uint16_t)f2bf(silu_f(c[r]));
    }
  }
  __syncthreads();

  // layer 2: @ Wh2 -> dh ; h = (h + dh) * mask
  if (active) {
    v16bf af[2];
#pragma unroll
    for (int ks = 0; ks < 2; ++ks) af[ks] = load_a_frag(L.m1, DH, ks, lane);
#pragma unroll
    for (int ns = 0; ns < 4; ++ns) {
      float bias = bh2[step*DH + ns*16 + colc];
      v8f c;
#pragma unroll
      for (int r = 0; r < 8; ++r) c[r] = bias;
#pragma unroll
      for (int ks = 0; ks < 2; ++ks)
        c = wmma_bf16(af[ks], load_b_frag(pws + (OFF_WH2 + (ks*4+ns)*512)/2, lane), c);
#pragma unroll
      for (int r = 0; r < 8; ++r) {
        int row = row0 + r;
        int n = n0 + row;
        if (n < NPTS) {
          size_t idx = ((size_t)b*NPTS + n)*DH + ns*16 + colc;
          h[idx] = (h[idx] + c[r]) * L.nmask[row];
        }
      }
    }
    // equivariant coordinate update: x += (dx_sum / K) * mask
    if (lane < 16) {
      int n = n0 + lane;
      if (n < NPTS) {
        float mv = mask[b*NPTS + n] * (1.f / (float)KNN);
#pragma unroll
        for (int d = 0; d < SPD; ++d) {
          size_t xi = ((size_t)b*NPTS + n)*SPD + d;
          x[xi] += dxb[xi] * mv;
        }
      }
    }
  }
}

// ---------------- workspace layout ----------------
static constexpr size_t aup(size_t v) { return (v + 255) & ~(size_t)255; }
static constexpr size_t OFF_COND = 0;
static constexpr size_t OFF_X    = aup(OFF_COND + (size_t)BATCH*DC*4);
static constexpr size_t OFF_H    = aup(OFF_X    + (size_t)BATCH*NPTS*SPD*4);
static constexpr size_t OFF_AGG  = aup(OFF_H    + (size_t)BATCH*NPTS*DH*4);
static constexpr size_t OFF_DXB  = aup(OFF_AGG  + (size_t)BATCH*NPTS*DH*4);
static constexpr size_t OFF_TGT  = aup(OFF_DXB  + (size_t)BATCH*NPTS*SPD*4);
static constexpr size_t OFF_PW   = aup(OFF_TGT  + (size_t)BATCH*NEDGE*4);

extern "C" void kernel_launch(void* const* d_in, const int* in_sizes, int n_in,
                              void* d_out, int out_size, void* d_ws, size_t ws_size,
                              hipStream_t stream) {
  const float* z    = (const float*)d_in[0];
  const float* t    = (const float*)d_in[1];
  const float* cnd  = (const float*)d_in[2];
  const float* mask = (const float*)d_in[3];
  const float* Wc1  = (const float*)d_in[4];
  const float* bc1  = (const float*)d_in[5];
  const float* Wc2  = (const float*)d_in[6];
  const float* bc2  = (const float*)d_in[7];
  const float* Wc3  = (const float*)d_in[8];
  const float* bc3  = (const float*)d_in[9];
  const float* Wemb = (const float*)d_in[10];
  const float* bemb = (const float*)d_in[11];
  const float* We1  = (const float*)d_in[12];
  const float* be1  = (const float*)d_in[13];
  const float* We2  = (const float*)d_in[14];
  const float* be2  = (const float*)d_in[15];
  const float* Wx1  = (const float*)d_in[16];
  const float* bx1  = (const float*)d_in[17];
  const float* Wx2  = (const float*)d_in[18];
  const float* bx2  = (const float*)d_in[19];
  const float* Wh1  = (const float*)d_in[20];
  const float* bh1  = (const float*)d_in[21];
  const float* Wh2  = (const float*)d_in[22];
  const float* bh2  = (const float*)d_in[23];

  char* ws = (char*)d_ws;
  float*    condW = (float*)(ws + OFF_COND);
  float*    x     = (float*)(ws + OFF_X);
  float*    h     = (float*)(ws + OFF_H);
  float*    agg   = (float*)(ws + OFF_AGG);
  float*    dxb   = (float*)(ws + OFF_DXB);
  int*      tgt   = (int*)(ws + OFF_TGT);
  uint32_t* pw    = (uint32_t*)(ws + OFF_PW);
  float*    out   = (float*)d_out;

  pack_weights_kernel<<<dim3((TSTEPS*68*32 + 255)/256), 256, 0, stream>>>(We1, We2, Wx1, Wh1, Wh2, pw);
  cond_kernel<<<dim3(1), 256, 0, stream>>>(t, cnd, Wc1, bc1, Wc2, bc2, Wc3, bc3, condW);
  knn_kernel<<<dim3((NPTS + KNN_T - 1)/KNN_T, BATCH), KNN_T, 0, stream>>>(z, mask, tgt);
  {
    int tot = BATCH*NPTS*DH + BATCH*NPTS*SPD;
    init_kernel<<<dim3((tot + 255)/256), 256, 0, stream>>>(z, Wemb, bemb, h, x);
  }
  for (int s = 0; s < TSTEPS; ++s) {
    int tot = BATCH*NPTS*DH + BATCH*NPTS*SPD;
    zero_kernel<<<dim3((tot + 255)/256), 256, 0, stream>>>(agg, dxb);
    edge_kernel<<<dim3((NTILE_E + WAVES - 1)/WAVES, BATCH), 32*WAVES, 0, stream>>>(
        x, h, mask, tgt, pw, be1, be2, bx1, Wx2, bx2, agg, dxb, s);
    node_kernel<<<dim3((NTILE_N + WAVES - 1)/WAVES, BATCH), 32*WAVES, 0, stream>>>(
        x, h, mask, agg, dxb, condW, pw, bh1, bh2, s);
  }
  out_kernel<<<dim3((BATCH*NPTS*SPD + 255)/256), 256, 0, stream>>>(z, mask, x, out);

  (void)in_sizes; (void)n_in; (void)out_size; (void)ws_size;
}
